// Head_13666585936313
// MI455X (gfx1250) — matrix operand order repaired and checked by reading, not compiled, not run on hardware
//
#include <hip/hip_runtime.h>
#include <hip/hip_bf16.h>

// Problem constants (from reference): B=8, T=2048, C=1024, H=64, fp32 in/out.
#define EMBED  1024
#define HEAD   64
#define BATCH  8
#define SEQ    2048
#define NTOK   (BATCH * SEQ)          // 16384 tokens
#define NTILE  (NTOK / 16)            // 1024 16-row tiles
#define KT_STEPS (EMBED / 32)         // 32 K-steps in projection

typedef __attribute__((ext_vector_type(16))) __bf16 v16bf;
typedef __attribute__((ext_vector_type(8)))  __bf16 v8bf;
typedef __attribute__((ext_vector_type(8)))  float  v8f;
typedef __attribute__((ext_vector_type(4)))  float  v4f;

#define NEG_BIG (-1.0e30f)

// Packed f32->bf16 conversion: one VALU op per 2 elements, result already in
// WMMA A-fragment dword layout (elem 2j in [15:0], 2j+1 in [31:16]).
__device__ __forceinline__ unsigned int cvt_pk_bf16(float lo, float hi) {
  unsigned int d;
  asm("v_cvt_pk_bf16_f32 %0, %1, %2" : "=v"(d) : "v"(lo), "v"(hi));
  return d;
}
__device__ __forceinline__ unsigned short f2bfbits(float f) {
  return (unsigned short)(cvt_pk_bf16(f, f) & 0xffffu);
}

typedef union {
  v16bf bf;
  unsigned int u32[8];
} afrag_t;

__device__ __forceinline__ v8f wmma_bf16(v16bf a, v16bf b, v8f c) {
  return __builtin_amdgcn_wmma_f32_16x16x32_bf16(false, a, false, b, (short)0, c,
                                                 false, false);
}
__device__ __forceinline__ void lds_fence() {
  asm volatile("s_wait_dscnt 0" ::: "memory");
}

// ---------------------------------------------------------------------------
// Kernel 1: repack Wq/Wk/Wv (fp32 [C][H]) into bf16 WMMA B-fragment order:
//   wf[mat][kt][nblk][lane][16]  where element i = W[kt*32 + (lane/16)*16 + i]
//                                                 [nblk*16 + (lane%16)]
// ---------------------------------------------------------------------------
__global__ void pack_w_kernel(const float* __restrict__ Wq,
                              const float* __restrict__ Wk,
                              const float* __restrict__ Wv,
                              unsigned short* __restrict__ wf) {
  int tid  = blockIdx.x * blockDim.x + threadIdx.x;   // 0..12287
  int lane = tid & 31;
  int nblk = (tid >> 5) & 3;
  int kt   = (tid >> 7) & 31;
  int mat  = tid >> 12;                               // 0..2
  const float* W = (mat == 0) ? Wq : ((mat == 1) ? Wk : Wv);
  int n = lane & 15, h = lane >> 4;
  unsigned short* dst = wf + ((((size_t)mat * KT_STEPS + kt) * 4 + nblk) * 32 + lane) * 16;
  int col  = nblk * 16 + n;
  int krow = kt * 32 + h * 16;
#pragma unroll
  for (int i = 0; i < 16; ++i)
    dst[i] = f2bfbits(W[(size_t)(krow + i) * HEAD + col]);
}

// ---------------------------------------------------------------------------
// Kernel 2: fused QKV projection. grid = 1024 tiles, 1 wave per block.
// Each wave: for a 16-token tile, D_qkv(3 x 16x64) = X(16x1024) @ W(1024x64).
// X is loaded + converted ONCE per tile; each A-frag feeds 12 WMMAs.
// Outputs: Q,K row-major bf16 [tok][64]; V transposed bf16 [b][h][t].
// ---------------------------------------------------------------------------
__global__ void __launch_bounds__(32)
qkv_proj_kernel(const float* __restrict__ x,
                const unsigned short* __restrict__ wf,
                unsigned short* __restrict__ qb,
                unsigned short* __restrict__ kb,
                unsigned short* __restrict__ vt) {
  const int tile = blockIdx.x;          // 0..1023
  const int lane = threadIdx.x;
  const int n = lane & 15, h = lane >> 4;
  const int tok0 = tile * 16;

  const v16bf* wfrag = (const v16bf*)wf;     // [mat][kt][nblk][32 lanes]
  const float* xrow  = x + (size_t)(tok0 + n) * EMBED;   // A-frag row m == lane&15

  v8f acc[3][4];
#pragma unroll
  for (int m = 0; m < 3; ++m)
#pragma unroll
    for (int nb = 0; nb < 4; ++nb) acc[m][nb] = v8f{};

  for (int kt = 0; kt < KT_STEPS; ++kt) {
    // A fragment: 16x32 bf16 of X. lane(m,h): elems 0..15 = cols kt*32+16h+i
    // (one contiguous 16-float segment).
    int c0 = kt * 32 + 16 * h;
    v4f xa0 = *(const v4f*)(xrow + c0);
    v4f xa1 = *(const v4f*)(xrow + c0 + 4);
    v4f xb0 = *(const v4f*)(xrow + c0 + 8);
    v4f xb1 = *(const v4f*)(xrow + c0 + 12);
    afrag_t A;
    A.u32[0] = cvt_pk_bf16(xa0[0], xa0[1]);
    A.u32[1] = cvt_pk_bf16(xa0[2], xa0[3]);
    A.u32[2] = cvt_pk_bf16(xa1[0], xa1[1]);
    A.u32[3] = cvt_pk_bf16(xa1[2], xa1[3]);
    A.u32[4] = cvt_pk_bf16(xb0[0], xb0[1]);
    A.u32[5] = cvt_pk_bf16(xb0[2], xb0[3]);
    A.u32[6] = cvt_pk_bf16(xb1[0], xb1[1]);
    A.u32[7] = cvt_pk_bf16(xb1[2], xb1[3]);
#pragma unroll
    for (int mat = 0; mat < 3; ++mat)
#pragma unroll
      for (int nblk = 0; nblk < 4; ++nblk) {
        v16bf b = wfrag[((size_t)mat * KT_STEPS + kt) * 4 * 32 + nblk * 32 + lane];
        acc[mat][nblk] = wmma_bf16(A.bf, b, acc[mat][nblk]);
      }
  }

  // Store D: lane(n,h), VGPR r -> row (8h+r), col (nblk*16+n)
#pragma unroll
  for (int nblk = 0; nblk < 4; ++nblk)
#pragma unroll
    for (int r = 0; r < 8; ++r) {
      int tok = tok0 + 8 * h + r;
      qb[(size_t)tok * HEAD + nblk * 16 + n] = f2bfbits(acc[0][nblk][r]);
      kb[(size_t)tok * HEAD + nblk * 16 + n] = f2bfbits(acc[1][nblk][r]);
    }
  {
    int bidx = tok0 >> 11;               // 2048 tokens per batch, tiles never straddle
    int tl0  = tok0 & (SEQ - 1);
#pragma unroll
    for (int nblk = 0; nblk < 4; ++nblk)
#pragma unroll
      for (int r = 0; r < 8; ++r) {
        int t    = tl0 + 8 * h + r;
        int feat = nblk * 16 + n;
        vt[((size_t)bidx * HEAD + feat) * SEQ + t] = f2bfbits(acc[2][nblk][r]);
      }
  }
}

// ---------------------------------------------------------------------------
// Kernel 3: causal flash attention. grid = 1024 (batch x 16-query tile),
// one wave per block. Online softmax, 64-key blocks, bf16 WMMA f32 accum.
// ---------------------------------------------------------------------------
__global__ void __launch_bounds__(32)
flash_attn_kernel(const unsigned short* __restrict__ qb,
                  const unsigned short* __restrict__ kb,
                  const unsigned short* __restrict__ vt,
                  float* __restrict__ out) {
  // 16 rows x 64 cols of P, rows padded to 72 bf16 (144B = 36 dwords stride;
  // gcd(36,64)=4 -> 16 row readers hit 16 distinct banks; 16B aligned chunks).
  __shared__ __align__(32) unsigned short Pld[16][72];

  const int qt   = blockIdx.x;          // global query tile 0..1023
  const int bidx = qt >> 7;             // 128 tiles per batch
  const int q0   = (qt & 127) * 16;     // local query start
  const int lane = threadIdx.x;
  const int n = lane & 15, h = lane >> 4;

  // Q A-fragments (16x32 bf16, two over H=64), row m == lane&15
  const unsigned short* qrow = qb + (size_t)(qt * 16 + n) * HEAD;
  v16bf aq[2];
#pragma unroll
  for (int kt = 0; kt < 2; ++kt) {
    v8bf s0 = *(const v8bf*)(qrow + kt * 32 + 16 * h);
    v8bf s1 = *(const v8bf*)(qrow + kt * 32 + 8 + 16 * h);
    v16bf a;
#pragma unroll
    for (int i = 0; i < 8; ++i) { a[i] = s0[i]; a[8 + i] = s1[i]; }
    aq[kt] = a;
  }

  float rm[8], rl[8];
#pragma unroll
  for (int r = 0; r < 8; ++r) { rm[r] = NEG_BIG; rl[r] = 0.0f; }
  v8f o[4] = {v8f{}, v8f{}, v8f{}, v8f{}};

  const unsigned short* kbb = kb + (size_t)bidx * SEQ * HEAD;
  const unsigned short* vtb = vt + (size_t)bidx * HEAD * SEQ;

  const int nblocks = (q0 + 15) / 64 + 1;      // 64-key blocks up to the diagonal
  for (int j = 0; j < nblocks; ++j) {
    const int k0 = j * 64;

    // Prefetch next K/V block while this one computes (global_prefetch_b8)
    if (j + 1 < nblocks) {
      __builtin_prefetch(kbb + (size_t)(k0 + 64 + lane) * HEAD, 0, 0);
      __builtin_prefetch(kbb + (size_t)(k0 + 96 + lane) * HEAD, 0, 0);
      __builtin_prefetch(vtb + (size_t)lane * SEQ + k0 + 64, 0, 0);
      __builtin_prefetch(vtb + (size_t)(lane + 32) * SEQ + k0 + 64, 0, 0);
    }

    // S(16x64) = Q(16x64) @ K^T(64x64): four 16x16 D tiles, K accum over kt.
    v8f s[4] = {v8f{}, v8f{}, v8f{}, v8f{}};
#pragma unroll
    for (int n0 = 0; n0 < 4; ++n0)
#pragma unroll
      for (int kt = 0; kt < 2; ++kt) {
        // B frag: b[i] = K[key k0+16*n0+n][feat kt*32+16h+i] -> contiguous 32B
        v16bf bk = *(const v16bf*)(kbb + (size_t)(k0 + n0 * 16 + n) * HEAD +
                                   kt * 32 + 16 * h);
        s[n0] = wmma_bf16(aq[kt], bk, s[n0]);
      }

    // Causal mask (only the trailing block(s) can need it)
    if (k0 + 63 > q0) {
#pragma unroll
      for (int n0 = 0; n0 < 4; ++n0)
#pragma unroll
        for (int r = 0; r < 8; ++r) {
          int key = k0 + n0 * 16 + n;
          int qi  = q0 + 8 * h + r;
          if (key > qi) s[n0][r] = NEG_BIG;
        }
    }

    // Online softmax per row (row lives across the 16 lanes of one half-wave)
#pragma unroll
    for (int r = 0; r < 8; ++r) {
      float mx = fmaxf(fmaxf(s[0][r], s[1][r]), fmaxf(s[2][r], s[3][r]));
#pragma unroll
      for (int d = 8; d >= 1; d >>= 1) mx = fmaxf(mx, __shfl_xor(mx, d, 32));
      float mnew  = fmaxf(rm[r], mx);
      float scale = __expf(rm[r] - mnew);     // 0 when rm == NEG_BIG
      rl[r] *= scale;
#pragma unroll
      for (int nblk = 0; nblk < 4; ++nblk) o[nblk][r] *= scale;
      float p[4], ps = 0.0f;
#pragma unroll
      for (int c = 0; c < 4; ++c) { p[c] = __expf(s[c][r] - mnew); ps += p[c]; }
#pragma unroll
      for (int d = 8; d >= 1; d >>= 1) ps += __shfl_xor(ps, d, 32);
      rl[r] += ps;
      rm[r]  = mnew;
      // Stage P (bf16) to LDS: D-frag layout -> A-frag layout transpose
#pragma unroll
      for (int c = 0; c < 4; ++c) Pld[8 * h + r][c * 16 + n] = f2bfbits(p[c]);
    }
    lds_fence();

    // Two A frags of P (16x64): frag kc, lane(m,h): elems 0..15 = P[m][kc*32+16h+i]
    v16bf ap[2];
#pragma unroll
    for (int kc = 0; kc < 2; ++kc) {
      v8bf p0v = *(const v8bf*)(&Pld[n][kc * 32 + 16 * h]);
      v8bf p1v = *(const v8bf*)(&Pld[n][kc * 32 + 8 + 16 * h]);
#pragma unroll
      for (int i = 0; i < 8; ++i) { ap[kc][i] = p0v[i]; ap[kc][8 + i] = p1v[i]; }
    }
    lds_fence();

    // O(16x64) += P(16x64) @ V(64x64): B frag from V^T is contiguous 32B
#pragma unroll
    for (int nblk = 0; nblk < 4; ++nblk)
#pragma unroll
      for (int kc = 0; kc < 2; ++kc) {
        v16bf bv = *(const v16bf*)(vtb + (size_t)(nblk * 16 + n) * SEQ +
                                   k0 + kc * 32 + 16 * h);
        o[nblk] = wmma_bf16(ap[kc], bv, o[nblk]);
      }
  }

  // Finalize: divide by row sum, store fp32
#pragma unroll
  for (int r = 0; r < 8; ++r) {
    float inv = 1.0f / rl[r];
    int tok = qt * 16 + 8 * h + r;
#pragma unroll
    for (int nblk = 0; nblk < 4; ++nblk)
      out[(size_t)tok * HEAD + nblk * 16 + n] = o[nblk][r] * inv;
  }
}

// ---------------------------------------------------------------------------
extern "C" void kernel_launch(void* const* d_in, const int* in_sizes, int n_in,
                              void* d_out, int out_size, void* d_ws, size_t ws_size,
                              hipStream_t stream) {
  const float* x  = (const float*)d_in[0];
  const float* Wq = (const float*)d_in[1];
  const float* Wk = (const float*)d_in[2];
  const float* Wv = (const float*)d_in[3];
  float* out = (float*)d_out;

  char* ws = (char*)d_ws;
  const size_t WF_BYTES  = (size_t)3 * KT_STEPS * 4 * 32 * 16 * 2;  // 393216
  const size_t QKV_BYTES = (size_t)NTOK * HEAD * 2;                 // 2 MB each
  unsigned short* wf = (unsigned short*)ws;
  unsigned short* qb = (unsigned short*)(ws + WF_BYTES);
  unsigned short* kb = (unsigned short*)(ws + WF_BYTES + QKV_BYTES);
  unsigned short* vt = (unsigned short*)(ws + WF_BYTES + 2 * QKV_BYTES);

  pack_w_kernel<<<48, 256, 0, stream>>>(Wq, Wk, Wv, wf);
  qkv_proj_kernel<<<NTILE, 32, 0, stream>>>(x, wf, qb, kb, vt);
  flash_attn_kernel<<<NTILE, 32, 0, stream>>>(qb, kb, vt, out);
}
